// EdgePredictor_1185410974254
// MI455X (gfx1250) — matrix operand order repaired and checked by reading, not compiled, not run on hardware
//
#include <hip/hip_runtime.h>
#include <hip/hip_bf16.h>

typedef float  v2f  __attribute__((ext_vector_type(2)));
typedef float  v8f  __attribute__((ext_vector_type(8)));
typedef __bf16 bf4v __attribute__((ext_vector_type(4)));
typedef __bf16 bf8v __attribute__((ext_vector_type(8)));
typedef __bf16 bf16v __attribute__((ext_vector_type(16)));

union Frag { bf16v f; bf8v h[2]; };

#define WMMA4(accv, av, bv) \
  accv = __builtin_amdgcn_wmma_f32_16x16x4_f32(false, av, false, bv, (short)0, accv, false, false)
#define WMMAB(accv, av, bv) \
  accv = __builtin_amdgcn_wmma_f32_16x16x32_bf16(false, av, false, bv, (short)0, accv, false, false)

__device__ __forceinline__ float elu(float x) {
    return x > 0.f ? x : (__expf(x) - 1.f);
}

struct BF2 { __bf16 hi, lo; };
__device__ __forceinline__ BF2 split_bf16(float x) {
    BF2 r;
    r.hi = (__bf16)x;
    r.lo = (__bf16)(x - (float)r.hi);
    return r;
}

// ---------------------------------------------------------------------------
// Kernel 1:  [U | V] = X @ W1   (M=2048, N=1024, K=256) -- full fp32 WMMA.
// Tiny (~1 GFLOP) thanks to the pair-factorization, so accuracy > speed here.
// ---------------------------------------------------------------------------
__global__ __launch_bounds__(256) void layer1_proj_kernel(
    const float* __restrict__ X,    // (2048, 256)
    const float* __restrict__ W1,   // (512, 512)
    float* __restrict__ U,          // (2048, 512)
    float* __restrict__ V)          // (2048, 512)
{
    const int lane = threadIdx.x & 31;
    const int wave = threadIdx.x >> 5;
    const int tile = blockIdx.x * 8 + wave;      // 0..8191
    const int tm   = tile >> 6;
    const int tn   = tile & 63;
    const int half = tn >> 5;                    // 0 -> U, 1 -> V
    const int ncol = ((tn & 31) << 4) + (lane & 15);
    const int koff = (lane < 16) ? 0 : 2;
    const int arow = (tm << 4) + (lane & 15);

    const float* Wb = W1 + half * 256 * 512;
    const float* Xr = X + arow * 256;

    v8f acc = {};
    #pragma unroll 4
    for (int k = 0; k < 256; k += 4) {
        v2f a = *(const v2f*)(Xr + k + koff);
        v2f b;
        b.x = Wb[(k + koff) * 512 + ncol];
        b.y = Wb[(k + koff + 1) * 512 + ncol];
        WMMA4(acc, a, b);
    }

    float* Ob = half ? V : U;
    const int rbase = (tm << 4) + ((lane >> 4) << 3);
    #pragma unroll
    for (int i = 0; i < 8; ++i)
        Ob[(rbase + i) * 512 + ncol] = acc[i];
}

// ---------------------------------------------------------------------------
// Kernel 2: 32 pairs x 2 orders = 64 rows per block.
//   h1 = elu(U[na]+V[nb]+b1)  -> bf16 hi/lo in LDS (persistent, K=512)
//   h2 = elu(h1 @ W2 + b2)    -> bf16x3 WMMA, W2 chunks double-buffered in LDS
//   out = 0.5*(h2f0@W3 + h2f1@W3) + b3
// ---------------------------------------------------------------------------
#define NPAIRS 2016
#define KC     32            // K chunk (= one bf16 WMMA K)
#define NCH    16            // 512 / KC
#define H1S    520           // ushort stride of h1 hi/lo rows (pad: bank spread)
#define W2S    40            // ushort stride per n-row of a W2 chunk (KC + 8)
#define H2S    516           // float stride of h2 rows

// LDS byte offsets
#define H1_BYTES   (2u * 64 * H1S * 2)          // hi + lo = 133120
#define W2BUF_USH  (2u * 512 * W2S)             // hi + lo, one buffer (ushorts)
#define W2_BYTES   (2u * W2BUF_USH * 2)         // two buffers = 163840
#define SMEM_BYTES (H1_BYTES + W2_BYTES + 64 * 4)

__device__ __forceinline__ void stage_load(const float* __restrict__ W2,
                                           int c, int t, float4 g[8]) {
    // thread t owns two 4x4 blocks: (kb, nb) and (kb+4, nb)
    const int kb = t >> 7, nb = t & 127;
    const float* base = W2 + (c * KC + kb * 4) * 512 + nb * 4;
    #pragma unroll
    for (int j = 0; j < 4; ++j) {
        g[j]     = *(const float4*)(base + j * 512);
        g[4 + j] = *(const float4*)(base + 16 * 512 + j * 512);
    }
}

__device__ __forceinline__ void stage_store(ushort* __restrict__ w2th,
                                            ushort* __restrict__ w2tl,
                                            int t, const float4 g[8]) {
    const int kb = t >> 7, nb = t & 127;
    #pragma unroll
    for (int b = 0; b < 2; ++b) {
        const int k0 = kb * 4 + b * 16;
        #pragma unroll
        for (int j = 0; j < 4; ++j) {        // column n = nb*4 + j
            bf4v hv, lv;
            #pragma unroll
            for (int kk = 0; kk < 4; ++kk) {
                const float x = ((const float*)&g[b * 4 + kk])[j];
                const BF2 s = split_bf16(x);
                hv[kk] = s.hi; lv[kk] = s.lo;
            }
            const int n = nb * 4 + j;
            *(bf4v*)(w2th + n * W2S + k0) = hv;   // 8B-aligned ds_store_b64
            *(bf4v*)(w2tl + n * W2S + k0) = lv;
        }
    }
}

__global__ __launch_bounds__(512) void mlp_pair_kernel(
    const float* __restrict__ U,
    const float* __restrict__ V,
    const float* __restrict__ b1,
    const float* __restrict__ W2,    // (512, 512)
    const float* __restrict__ b2,
    const float* __restrict__ W3,    // (512,)
    const float* __restrict__ b3,
    float* __restrict__ out)         // (64512,)
{
    extern __shared__ char smem[];
    ushort* h1h   = (ushort*)smem;                 // [64][520] bf16 hi
    ushort* h1l   = h1h + 64 * H1S;                // [64][520] bf16 lo
    float*  h2    = (float*)smem;                  // overlays h1 after GEMM
    ushort* w2buf = (ushort*)(smem + H1_BYTES);    // 2 x (hi[512*40] lo[512*40])
    float*  rsum  = (float*)(smem + H1_BYTES + W2_BYTES);

    const int tid  = threadIdx.x;
    const int lane = tid & 31;
    const int wave = tid >> 5;
    const int blk  = blockIdx.x;

    // ---- Phase 0: assemble h1 rows, split to bf16 hi/lo ----------------
    {
        const int r     = tid >> 3;
        const int seg   = tid & 7;
        const int gp    = blk * 32 + (r >> 1);
        const int order = r & 1;
        const int bidx  = gp / NPAIRS;
        int p = gp - bidx * NPAIRS;
        int ia = 0, cnt = 63;
        while (p >= cnt) { p -= cnt; --cnt; ++ia; }
        const int jb = ia + 1 + p;
        const int na = order ? jb : ia;
        const int nb = order ? ia : jb;
        const float* ur = U + (bidx * 64 + na) * 512;
        const float* vr = V + (bidx * 64 + nb) * 512;
        ushort* hh = h1h + r * H1S;
        ushort* hl = h1l + r * H1S;
        const int c0 = seg * 64;
        for (int c = c0; c < c0 + 64; c += 4) {
            float4 uu = *(const float4*)(ur + c);
            float4 vv = *(const float4*)(vr + c);
            float4 bb = *(const float4*)(b1 + c);
            const float x0 = elu(uu.x + vv.x + bb.x);
            const float x1 = elu(uu.y + vv.y + bb.y);
            const float x2 = elu(uu.z + vv.z + bb.z);
            const float x3 = elu(uu.w + vv.w + bb.w);
            bf4v hv, lv;
            const BF2 s0 = split_bf16(x0); hv[0] = s0.hi; lv[0] = s0.lo;
            const BF2 s1 = split_bf16(x1); hv[1] = s1.hi; lv[1] = s1.lo;
            const BF2 s2 = split_bf16(x2); hv[2] = s2.hi; lv[2] = s2.lo;
            const BF2 s3 = split_bf16(x3); hv[3] = s3.hi; lv[3] = s3.lo;
            *(bf4v*)(hh + c) = hv;
            *(bf4v*)(hl + c) = lv;
        }
    }

    // ---- prologue of W2 pipeline: stage chunk 0 -------------------------
    float4 g[8];
    stage_load(W2, 0, tid, g);
    stage_store(w2buf, w2buf + 512 * W2S, tid, g);
    __syncthreads();

    // ---- Phase 1: h2 = h1 @ W2 via bf16x3 WMMA, double-buffered ---------
    const int rg  = wave & 1;          // row group: rows [rg*32, rg*32+32)
    const int cg  = wave >> 1;         // col group: cols [cg*64, cg*64+64)
    const int ml  = lane & 15;
    const int ksa = (lane < 16) ? 0 : 8;
    const int ksb = (lane < 16) ? 0 : 8;

    v8f acc[2][4];
    #pragma unroll
    for (int ri = 0; ri < 2; ++ri)
        #pragma unroll
        for (int j = 0; j < 4; ++j) acc[ri][j] = (v8f){};

    for (int c = 0; c < NCH; ++c) {
        float4 gn[8];
        if (c < NCH - 1) stage_load(W2, c + 1, tid, gn);

        const ushort* w2th = w2buf + (c & 1) * 2 * 512 * W2S;
        const ushort* w2tl = w2th + 512 * W2S;
        const int kabs = c * KC;

        Frag aH[2], aL[2];
        #pragma unroll
        for (int ri = 0; ri < 2; ++ri) {
            const int row = (rg * 2 + ri) * 16 + ml;
            const ushort* ph = h1h + row * H1S + kabs + ksa;
            const ushort* pl = h1l + row * H1S + kabs + ksa;
            aH[ri].h[0] = *(const bf8v*)(ph);
            aH[ri].h[1] = *(const bf8v*)(ph + 16);
            aL[ri].h[0] = *(const bf8v*)(pl);
            aL[ri].h[1] = *(const bf8v*)(pl + 16);
        }
        #pragma unroll
        for (int j = 0; j < 4; ++j) {
            const int n = (cg * 4 + j) * 16 + ml;
            const ushort* ph = w2th + n * W2S + ksb;
            const ushort* pl = w2tl + n * W2S + ksb;
            Frag bH, bL;
            bH.h[0] = *(const bf8v*)(ph);  bH.h[1] = *(const bf8v*)(ph + 16);
            bL.h[0] = *(const bf8v*)(pl);  bL.h[1] = *(const bf8v*)(pl + 16);
            #pragma unroll
            for (int ri = 0; ri < 2; ++ri) {
                WMMAB(acc[ri][j], aH[ri].f, bH.f);
                WMMAB(acc[ri][j], aH[ri].f, bL.f);
                WMMAB(acc[ri][j], aL[ri].f, bH.f);
            }
        }

        if (c < NCH - 1) {
            ushort* dsth = w2buf + ((c + 1) & 1) * 2 * 512 * W2S;
            stage_store(dsth, dsth + 512 * W2S, tid, gn);
        }
        __syncthreads();   // buffers swap; also final: h1 reads done
    }

    // ---- Phase 2: bias + ELU, h2 overwrites h1 region -------------------
    {
        const int rb = (lane >> 4) << 3;
        #pragma unroll
        for (int ri = 0; ri < 2; ++ri) {
            const int rbase = (rg * 2 + ri) * 16 + rb;
            #pragma unroll
            for (int j = 0; j < 4; ++j) {
                const int n = (cg * 4 + j) * 16 + ml;
                const float bias = b2[n];
                #pragma unroll
                for (int i = 0; i < 8; ++i)
                    h2[(rbase + i) * H2S + n] = elu(acc[ri][j][i] + bias);
            }
        }
    }
    __syncthreads();

    // ---- Phase 3: h2 @ W3, combine f0/f1 --------------------------------
    {
        const int r   = tid >> 3;
        const int seg = tid & 7;
        const float* hr = h2 + r * H2S;
        float s = 0.f;
        #pragma unroll 8
        for (int cc = seg * 64; cc < seg * 64 + 64; ++cc)
            s += hr[cc] * W3[cc];
        s += __shfl_xor(s, 1, 32);
        s += __shfl_xor(s, 2, 32);
        s += __shfl_xor(s, 4, 32);
        if (seg == 0) rsum[r] = s;
    }
    __syncthreads();

    if (tid < 32)
        out[blk * 32 + tid] = 0.5f * (rsum[2 * tid] + rsum[2 * tid + 1]) + b3[0];
}

// ---------------------------------------------------------------------------
extern "C" void kernel_launch(void* const* d_in, const int* in_sizes, int n_in,
                              void* d_out, int out_size, void* d_ws, size_t ws_size,
                              hipStream_t stream) {
    const float* x  = (const float*)d_in[0];   // (32, 64, 256)
    const float* W1 = (const float*)d_in[1];   // (512, 512)
    const float* b1 = (const float*)d_in[2];   // (512,)
    const float* W2 = (const float*)d_in[3];   // (512, 512)
    const float* b2 = (const float*)d_in[4];   // (512,)
    const float* W3 = (const float*)d_in[5];   // (512, 1)
    const float* b3 = (const float*)d_in[6];   // (1,)

    float* Uw = (float*)d_ws;                  // 2048*512 floats
    float* Vw = Uw + 2048 * 512;               // 2048*512 floats (8 MB total)
    float* outp = (float*)d_out;               // 64512 floats

    layer1_proj_kernel<<<1024, 256, 0, stream>>>(x, W1, Uw, Vw);
    mlp_pair_kernel<<<2016, 512, SMEM_BYTES, stream>>>(Uw, Vw, b1, W2, b2, W3,
                                                       b3, outp);
}
